// Model_7438883356780
// MI455X (gfx1250) — matrix operand order repaired and checked by reading, not compiled, not run on hardware
//
#include <hip/hip_runtime.h>
#include <math.h>

#define BATCH  32
#define SEQL   512
#define DMODEL 512
#define NHEADS 8
#define EHEAD  64
#define DFF    2048
#define CIN    21
#define MARKN  4
#define PREDL  256
#define NROWS  (BATCH*SEQL)   // 16384 rows for both enc and dec streams

typedef __attribute__((ext_vector_type(16))) __bf16 v16bf;
typedef __attribute__((ext_vector_type(8)))  __bf16 v8bf;
typedef __attribute__((ext_vector_type(4)))  __bf16 v4bf;
typedef __attribute__((ext_vector_type(8)))  float  v8f;

struct alignas(16) F4 { float x, y, z, w; };
union Frag16 { v16bf v; v8bf h[2]; };

// ---------------------------------------------------------------------------
// Generic bf16 WMMA GEMM:  C[M,N] = act(A[M,K] * B + bias)
//   A fp32 row-major (lda).  B fp32: transB==0 -> [K,N] (ldb), transB==1 -> [N,K] (ldb)
//   Batched via blockIdx.z decomposed as (b = z/Hs, h = z%Hs) with separate strides.
//   Workgroup tile 128x128, 8 waves (4 in M x 2 in N), wave tile 32x64,
//   K-step 32 using V_WMMA_F32_16X16X32_BF16, double-buffered LDS so next-tile
//   global loads overlap the WMMA block.
//   Requirements (all callers satisfy): M % 128 == 0, K % 32 == 0, N % 4 == 0.
// ---------------------------------------------------------------------------
__global__ __launch_bounds__(256) void k_gemm_bf16(
    const float* __restrict__ A, const float* __restrict__ Bm,
    const float* __restrict__ bias, float* __restrict__ Co,
    int M, int N, int K, int lda, int ldb, int ldc,
    int transB, int act, int Hs,
    long long sAb, long long sAh, long long sBb, long long sBh,
    long long sCb, long long sCh)
{
    constexpr int TM = 128, TN = 128, TK = 32, LSTR = 40; // LDS stride in bf16 units
    __shared__ __align__(16) __bf16 As[2][TM * LSTR];
    __shared__ __align__(16) __bf16 Bs[2][TN * LSTR];

    const int z  = blockIdx.z, bz = z / Hs, hz = z % Hs;
    A  += (long long)bz * sAb + (long long)hz * sAh;
    Bm += (long long)bz * sBb + (long long)hz * sBh;
    Co += (long long)bz * sCb + (long long)hz * sCh;

    const int tM = blockIdx.y * TM, tN = blockIdx.x * TN;
    const int tid = threadIdx.x, lane = tid & 31, wid = tid >> 5;
    const int wM = (wid & 3) * 32, wN = (wid >> 2) * 64;
    const int half = lane >> 4, lr = lane & 15;

    v8f acc[2][4];
    const v8f vzero = {0.f,0.f,0.f,0.f,0.f,0.f,0.f,0.f};
    for (int i = 0; i < 2; ++i)
        for (int j = 0; j < 4; ++j) acc[i][j] = vzero;

    // ---- tile staging helpers (registers <-> LDS, fp32 -> bf16) ----
    auto loadA = [&](int k0, F4* ra) {
#pragma unroll
        for (int u = 0; u < 4; ++u) {
            int i = tid + u * 256;
            int r = i >> 3, c4 = (i & 7) << 2;            // 8 float4 per 32-wide row
            ra[u] = *(const F4*)(A + (long long)(tM + r) * lda + (k0 + c4));
        }
    };
    auto loadB = [&](int k0, F4* rb) {
        if (transB) {
#pragma unroll
            for (int u = 0; u < 4; ++u) {
                int i = tid + u * 256;
                int n = i >> 3, c4 = (i & 7) << 2;
                int gn = tN + n;
                F4 v = {0.f,0.f,0.f,0.f};
                if (gn < N) v = *(const F4*)(Bm + (long long)gn * ldb + (k0 + c4));
                rb[u] = v;
            }
        } else {
#pragma unroll
            for (int u = 0; u < 4; ++u) {
                int i = tid + u * 256;
                int kk = i >> 5, n4 = (i & 31) << 2;      // 32 float4 per 128-wide row
                int gn = tN + n4;
                F4 v = {0.f,0.f,0.f,0.f};
                if (gn < N) v = *(const F4*)(Bm + (long long)(k0 + kk) * ldb + gn);
                rb[u] = v;
            }
        }
    };
    auto storeA = [&](int buf, const F4* ra) {
#pragma unroll
        for (int u = 0; u < 4; ++u) {
            int i = tid + u * 256;
            int r = i >> 3, c4 = (i & 7) << 2;
            v4bf p;
            p[0] = (__bf16)ra[u].x; p[1] = (__bf16)ra[u].y;
            p[2] = (__bf16)ra[u].z; p[3] = (__bf16)ra[u].w;
            *(v4bf*)&As[buf][r * LSTR + c4] = p;          // ds_store_b64
        }
    };
    auto storeB = [&](int buf, const F4* rb) {
        if (transB) {
#pragma unroll
            for (int u = 0; u < 4; ++u) {
                int i = tid + u * 256;
                int n = i >> 3, c4 = (i & 7) << 2;
                v4bf p;
                p[0] = (__bf16)rb[u].x; p[1] = (__bf16)rb[u].y;
                p[2] = (__bf16)rb[u].z; p[3] = (__bf16)rb[u].w;
                *(v4bf*)&Bs[buf][n * LSTR + c4] = p;
            }
        } else {
#pragma unroll
            for (int u = 0; u < 4; ++u) {                 // transpose scatter Bs[n][k]
                int i = tid + u * 256;
                int kk = i >> 5, n4 = (i & 31) << 2;
                Bs[buf][(n4 + 0) * LSTR + kk] = (__bf16)rb[u].x;
                Bs[buf][(n4 + 1) * LSTR + kk] = (__bf16)rb[u].y;
                Bs[buf][(n4 + 2) * LSTR + kk] = (__bf16)rb[u].z;
                Bs[buf][(n4 + 3) * LSTR + kk] = (__bf16)rb[u].w;
            }
        }
    };

    // ---- prologue: stage tile 0 ----
    F4 ra[4], rb[4];
    loadA(0, ra); loadB(0, rb);
    storeA(0, ra); storeB(0, rb);
    __syncthreads();

    const int nsteps = K / TK;
    for (int s = 0; s < nsteps; ++s) {
        const int buf = s & 1;
        // issue next tile's global loads before the math so they overlap WMMA
        if (s + 1 < nsteps) {
            loadA((s + 1) * TK, ra);
            loadB((s + 1) * TK, rb);
        }
        // ---- fragments + WMMA ----
        // A frag (16-bit A 16x32): lane half 0 -> K 0..7 & 16..23, half 1 -> K 8..15 & 24..31
        Frag16 fa[2];
#pragma unroll
        for (int i = 0; i < 2; ++i) {
            const __bf16* ap = &As[buf][(wM + i * 16 + lr) * LSTR + half * 8];
            fa[i].h[0] = *(const v8bf*)(ap);
            fa[i].h[1] = *(const v8bf*)(ap + 16);
        }
#pragma unroll
        for (int j = 0; j < 4; ++j) {
            Frag16 fb;  // B frag: lane half 0 -> K 0..7 & 16..23, half 1 -> K 8..15 & 24..31 (col = lr)
            const __bf16* bp = &Bs[buf][(wN + j * 16 + lr) * LSTR + half * 8];
            fb.h[0] = *(const v8bf*)(bp);
            fb.h[1] = *(const v8bf*)(bp + 16);
#pragma unroll
            for (int i = 0; i < 2; ++i)
                acc[i][j] = __builtin_amdgcn_wmma_f32_16x16x32_bf16(
                    false, fa[i].v, false, fb.v, (short)0, acc[i][j], false, false);
        }
        __syncthreads();           // buf^1 readers (prev step) done before overwrite
        if (s + 1 < nsteps) {
            storeA(buf ^ 1, ra);
            storeB(buf ^ 1, rb);
            __syncthreads();       // buf^1 fully written before next step reads it
        }
    }

    // ---- epilogue: C layout VGPR r, lane<16 -> M=r, lane>=16 -> M=8+r, N=lane%16 ----
    for (int j = 0; j < 4; ++j) {
        int col = tN + wN + j * 16 + lr;
        if (col >= N) continue;
        float bv = bias ? bias[col] : 0.f;
#pragma unroll
        for (int i = 0; i < 2; ++i) {
            int rb0 = tM + wM + i * 16 + half * 8;
#pragma unroll
            for (int r = 0; r < 8; ++r) {
                float x = acc[i][j][r] + bv;
                if (act == 1) x = 0.5f * x * (1.f + erff(x * 0.70710678118f)); // exact GELU
                Co[(long long)(rb0 + r) * ldc + col] = x;
            }
        }
    }
}

// ---------------------------------------------------------------------------
// Per-(b,c) mean/std over time
// ---------------------------------------------------------------------------
__global__ void k_stats(const float* __restrict__ x, float* __restrict__ meanv,
                        float* __restrict__ stdv)
{
    int bc = blockIdx.x, b = bc / CIN, c = bc % CIN, tid = threadIdx.x;
    __shared__ float s1[64], s2[64];
    float a = 0.f, a2 = 0.f;
    for (int l = tid; l < SEQL; l += 64) {
        float v = x[((long long)b * SEQL + l) * CIN + c];
        a += v; a2 += v * v;
    }
    s1[tid] = a; s2[tid] = a2; __syncthreads();
    for (int st = 32; st > 0; st >>= 1) {
        if (tid < st) { s1[tid] += s1[tid + st]; s2[tid] += s2[tid + st]; }
        __syncthreads();
    }
    if (tid == 0) {
        float m = s1[0] * (1.f / SEQL);
        float var = s2[0] * (1.f / SEQL) - m * m;
        meanv[bc] = m;
        stdv[bc]  = sqrtf(var + 1e-5f);
    }
}

__global__ void k_norm(const float* __restrict__ x, const float* __restrict__ meanv,
                       const float* __restrict__ stdv, float* __restrict__ xn, int tot)
{
    int idx = blockIdx.x * 256 + threadIdx.x;
    if (idx >= tot) return;
    int c = idx % CIN, b = idx / (SEQL * CIN);
    xn[idx] = (x[idx] - meanv[b * CIN + c]) / stdv[b * CIN + c];
}

__global__ void k_decnew(const float* __restrict__ xn, float* __restrict__ xd, int tot)
{
    int idx = blockIdx.x * 256 + threadIdx.x;
    if (idx >= tot) return;
    int l = (idx / CIN) % SEQL;
    xd[idx] = (l < SEQL - PREDL) ? xn[idx + PREDL * CIN] : 0.f;
}

// ---------------------------------------------------------------------------
// Projector: series conv (circular over channel axis) + concat stats
// ---------------------------------------------------------------------------
__global__ void k_pconv(const float* __restrict__ x, const float* __restrict__ w,
                        const float* __restrict__ st, float* __restrict__ cat)
{
    int b = blockIdx.x, e = threadIdx.x;
    if (e >= CIN) return;
    int em = (e + CIN - 1) % CIN, ep = (e + 1) % CIN;
    const float* xb = x + (long long)b * SEQL * CIN;
    float acc = 0.f;
    for (int l = 0; l < SEQL; ++l) {
        const float* xr = xb + l * CIN;
        acc += xr[em] * w[l * 3 + 0] + xr[e] * w[l * 3 + 1] + xr[ep] * w[l * 3 + 2];
    }
    cat[b * 2 * CIN + e]       = acc;
    cat[b * 2 * CIN + CIN + e] = st[b * CIN + e];
}

__global__ void k_mlp(const float* __restrict__ in, const float* __restrict__ w,
                      const float* __restrict__ bias, float* __restrict__ out,
                      int din, int dout, int relu)
{
    int b = blockIdx.x;
    __shared__ float xi[128];
    if ((int)threadIdx.x < din) xi[threadIdx.x] = in[b * din + threadIdx.x];
    __syncthreads();
    for (int j = threadIdx.x; j < dout; j += blockDim.x) {
        float acc = bias ? bias[j] : 0.f;
        for (int i = 0; i < din; ++i) acc += xi[i] * w[i * dout + j];
        if (relu) acc = fmaxf(acc, 0.f);
        out[b * dout + j] = acc;
    }
}

__global__ void k_exp(const float* __restrict__ in, float* __restrict__ out)
{
    int b = threadIdx.x;
    if (b < BATCH) out[b] = __expf(in[b]);
}

// ---------------------------------------------------------------------------
// Token embedding (circular conv over time, k=3) + positional + mark embedding
// ---------------------------------------------------------------------------
__global__ __launch_bounds__(128) void k_embed(
    const float* __restrict__ x, const float* __restrict__ mark,
    const float* __restrict__ cw, const float* __restrict__ mw,
    float* __restrict__ out)
{
    int row = blockIdx.x, b = row >> 9, l = row & 511;
    int lm = (l + SEQL - 1) & 511, lp = (l + 1) & 511;
    __shared__ float xs[3 * CIN + MARKN];
    int tid = threadIdx.x;
    if (tid < CIN) {
        xs[tid]           = x[((long long)b * SEQL + lm) * CIN + tid];
        xs[CIN + tid]     = x[((long long)b * SEQL + l ) * CIN + tid];
        xs[2 * CIN + tid] = x[((long long)b * SEQL + lp) * CIN + tid];
    }
    if (tid < MARKN) xs[3 * CIN + tid] = mark[((long long)b * SEQL + l) * MARKN + tid];
    __syncthreads();
    for (int d = tid; d < DMODEL; d += 128) {
        float freq = __expf(-(float)(d & ~1) * 0.0179889460399f); // ln(1e4)/512
        float ang  = (float)l * freq;
        float acc  = (d & 1) ? __cosf(ang) : __sinf(ang);
        for (int c = 0; c < CIN; ++c)
            acc += xs[c]           * cw[(0 * CIN + c) * DMODEL + d]
                 + xs[CIN + c]     * cw[(1 * CIN + c) * DMODEL + d]
                 + xs[2 * CIN + c] * cw[(2 * CIN + c) * DMODEL + d];
        for (int m = 0; m < MARKN; ++m) acc += xs[3 * CIN + m] * mw[m * DMODEL + d];
        out[(long long)row * DMODEL + d] = acc;
    }
}

// ---------------------------------------------------------------------------
// De-stationary softmax: softmax(scale*(scores*tau[b] + delta[b,s])), opt causal
// ---------------------------------------------------------------------------
__global__ __launch_bounds__(256) void k_softmax(
    float* __restrict__ S, const float* __restrict__ tauv,
    const float* __restrict__ deltav, float scale, int causal)
{
    int l = blockIdx.x, z = blockIdx.y, b = z >> 3; // NHEADS==8
    int tid = threadIdx.x;
    float* row = S + ((long long)z * SEQL + l) * SEQL;
    float ts = tauv[b] * scale;
    float v0 = row[tid] * ts, v1 = row[tid + 256] * ts;
    if (deltav) {
        v0 += deltav[b * SEQL + tid] * scale;
        v1 += deltav[b * SEQL + tid + 256] * scale;
    }
    if (causal) {
        if (tid > l)       v0 = -1e9f;
        if (tid + 256 > l) v1 = -1e9f;
    }
    __shared__ float red[256];
    red[tid] = fmaxf(v0, v1); __syncthreads();
    for (int st = 128; st > 0; st >>= 1) {
        if (tid < st) red[tid] = fmaxf(red[tid], red[tid + st]);
        __syncthreads();
    }
    float m = red[0]; __syncthreads();
    float e0 = __expf(v0 - m), e1 = __expf(v1 - m);
    red[tid] = e0 + e1; __syncthreads();
    for (int st = 128; st > 0; st >>= 1) {
        if (tid < st) red[tid] += red[tid + st];
        __syncthreads();
    }
    float inv = 1.f / red[0];
    row[tid] = e0 * inv; row[tid + 256] = e1 * inv;
}

// ---------------------------------------------------------------------------
// out = LayerNorm(X (+ R)) * w + b     (row length 512, in-place safe)
// ---------------------------------------------------------------------------
__global__ __launch_bounds__(256) void k_add_ln(
    const float* __restrict__ X, const float* __restrict__ R,
    const float* __restrict__ w, const float* __restrict__ b,
    float* __restrict__ O)
{
    __shared__ float s1[256], s2[256];
    int row = blockIdx.x, tid = threadIdx.x;
    const float* xr = X + (long long)row * DMODEL;
    float a0 = xr[tid], a1 = xr[tid + 256];
    if (R) {
        a0 += R[(long long)row * DMODEL + tid];
        a1 += R[(long long)row * DMODEL + tid + 256];
    }
    s1[tid] = a0 + a1; s2[tid] = a0 * a0 + a1 * a1; __syncthreads();
    for (int st = 128; st > 0; st >>= 1) {
        if (tid < st) { s1[tid] += s1[tid + st]; s2[tid] += s2[tid + st]; }
        __syncthreads();
    }
    float mean = s1[0] * (1.f / DMODEL);
    float var  = s2[0] * (1.f / DMODEL) - mean * mean;
    float rstd = rsqrtf(var + 1e-5f);
    float* orow = O + (long long)row * DMODEL;
    orow[tid]       = (a0 - mean) * rstd * w[tid] + b[tid];
    orow[tid + 256] = (a1 - mean) * rstd * w[tid + 256] + b[tid + 256];
}

// ---------------------------------------------------------------------------
// Final projection (N=21) fused with de-normalization + pred-window slice
// ---------------------------------------------------------------------------
__global__ void k_final(const float* __restrict__ Xd, const float* __restrict__ pw,
                        const float* __restrict__ pb, const float* __restrict__ meanv,
                        const float* __restrict__ stdv, float* __restrict__ out)
{
    int idx = blockIdx.x, b = idx >> 8, t = idx & 255, l = PREDL + t;
    int c = threadIdx.x;
    if (c >= CIN) return;
    const float* xr = Xd + ((long long)b * SEQL + l) * DMODEL;
    float acc = pb[c];
    for (int d = 0; d < DMODEL; ++d) acc += xr[d] * pw[d * CIN + c];
    out[((long long)b * PREDL + t) * CIN + c] = acc * stdv[b * CIN + c] + meanv[b * CIN + c];
}

// ===========================================================================
extern "C" void kernel_launch(void* const* d_in, const int* in_sizes, int n_in,
                              void* d_out, int out_size, void* d_ws, size_t ws_size,
                              hipStream_t stream)
{
    (void)in_sizes; (void)n_in; (void)out_size; (void)ws_size;
    auto IN = [&](int i) { return (const float*)d_in[i]; };
    const float* x_enc      = IN(0);
    const float* x_mark_enc = IN(1);
    const float* x_dec      = IN(2); (void)x_dec; // only shape/zeros used
    const float* x_mark_dec = IN(3);
    // Param leaf order (JAX pytree: sorted dict keys, lists in order, 'b' < 'w'):
    //  4 dec_emb.conv_w  5 dec_emb.mark_w
    //  dec_layers[0]: 6 cross.k.b 7 cross.k.w 8 cross.o.b 9 cross.o.w 10 cross.q.b
    //   11 cross.q.w 12 cross.v.b 13 cross.v.w 14 ff1.b 15 ff1.w 16 ff2.b 17 ff2.w
    //   18 ln1.b 19 ln1.w 20 ln2.b 21 ln2.w 22 ln3.b 23 ln3.w
    //   24 self.k.b 25 self.k.w 26 self.o.b 27 self.o.w 28 self.q.b 29 self.q.w
    //   30 self.v.b 31 self.v.w
    //  32 dec_ln.b 33 dec_ln.w
    //  delta: 34 conv_w 35 h0.b 36 h0.w 37 h1.b 38 h1.w 39 out_w
    //  40 enc_emb.conv_w 41 enc_emb.mark_w
    //  enc_layers[l] base 42+16l: +0 k.b +1 k.w +2 o.b +3 o.w +4 q.b +5 q.w +6 v.b
    //   +7 v.w +8 ff1.b +9 ff1.w +10 ff2.b +11 ff2.w +12 ln1.b +13 ln1.w +14 ln2.b +15 ln2.w
    //  74 enc_ln.b 75 enc_ln.w  76 proj.b 77 proj.w
    //  tau: 78 conv_w 79 h0.b 80 h0.w 81 h1.b 82 h1.w 83 out_w

    float* W = (float*)d_ws;
    size_t off = 0;
    auto alloc = [&](size_t n) { float* p = W + off; off += (n + 3) & ~(size_t)3; return p; };
    float* xn     = alloc((size_t)BATCH * SEQL * CIN);
    float* xdec   = alloc((size_t)BATCH * SEQL * CIN);
    float* meanv  = alloc(BATCH * CIN);
    float* stdv   = alloc(BATCH * CIN);
    float* cat    = alloc(BATCH * 2 * CIN);
    float* h0     = alloc(BATCH * 128);
    float* h1     = alloc(BATCH * 128);
    float* pout   = alloc(BATCH * SEQL);
    float* tauv   = alloc(BATCH);
    float* deltav = alloc(BATCH * SEQL);
    const size_t NR = (size_t)NROWS * DMODEL;
    float* X   = alloc(NR);
    float* Y   = alloc(NR);
    float* Qb  = alloc(NR);
    float* Kb  = alloc(NR);
    float* Vb  = alloc(NR);
    float* T1  = alloc(NR);
    float* ENC = alloc(NR);
    float* XD  = alloc(NR);
    float* FFH = alloc((size_t)NROWS * DFF);
    float* SC  = alloc((size_t)BATCH * NHEADS * SEQL * SEQL);

    auto gemm = [&](const float* A, const float* Bm, const float* bias, float* C,
                    int M, int N, int K, int lda, int ldb, int ldc, int transB, int act,
                    int Z, int Hs, long long sAb, long long sAh, long long sBb,
                    long long sBh, long long sCb, long long sCh) {
        dim3 g((N + 127) / 128, (M + 127) / 128, Z);
        k_gemm_bf16<<<g, 256, 0, stream>>>(A, Bm, bias, C, M, N, K, lda, ldb, ldc,
                                           transB, act, Hs, sAb, sAh, sBb, sBh, sCb, sCh);
    };
    auto dense = [&](const float* A, const float* Bm, const float* bias, float* C,
                     int M, int N, int K, int act) {
        gemm(A, Bm, bias, C, M, N, K, K, N, N, 0, act, 1, 1, 0, 0, 0, 0, 0, 0);
    };

    // ---- stats + normalization + decoder input ----
    k_stats<<<BATCH * CIN, 64, 0, stream>>>(x_enc, meanv, stdv);
    int tot = BATCH * SEQL * CIN;
    k_norm<<<(tot + 255) / 256, 256, 0, stream>>>(x_enc, meanv, stdv, xn, tot);
    k_decnew<<<(tot + 255) / 256, 256, 0, stream>>>(xn, xdec, tot);

    // ---- tau projector: exp(MLP([conv(x_raw), std])) ----
    k_pconv<<<BATCH, 32, 0, stream>>>(x_enc, IN(78), stdv, cat);
    k_mlp<<<BATCH, 128, 0, stream>>>(cat, IN(80), IN(79), h0, 2 * CIN, 128, 1);
    k_mlp<<<BATCH, 128, 0, stream>>>(h0, IN(82), IN(81), h1, 128, 128, 1);
    k_mlp<<<BATCH, 128, 0, stream>>>(h1, IN(83), nullptr, pout, 128, 1, 0);
    k_exp<<<1, 32, 0, stream>>>(pout, tauv);
    // ---- delta projector: MLP([conv(x_raw), mean]) -> [B,512] ----
    k_pconv<<<BATCH, 32, 0, stream>>>(x_enc, IN(34), meanv, cat);
    k_mlp<<<BATCH, 128, 0, stream>>>(cat, IN(36), IN(35), h0, 2 * CIN, 128, 1);
    k_mlp<<<BATCH, 128, 0, stream>>>(h0, IN(38), IN(37), h1, 128, 128, 1);
    k_mlp<<<BATCH, 128, 0, stream>>>(h1, IN(39), nullptr, deltav, 128, SEQL, 0);

    const long long LD = (long long)SEQL * DMODEL;   // per-batch activation stride
    const long long LSs = (long long)SEQL * SEQL;    // per-head score stride

    auto attention = [&](const float* Xq, const float* Xkv,
                         const float* qw, const float* qb, const float* kw, const float* kb,
                         const float* vw, const float* vb, const float* ow, const float* ob,
                         int causal, const float* dl, float* outY) {
        dense(Xq,  qw, qb, Qb, NROWS, DMODEL, DMODEL, 0);
        dense(Xkv, kw, kb, Kb, NROWS, DMODEL, DMODEL, 0);
        dense(Xkv, vw, vb, Vb, NROWS, DMODEL, DMODEL, 0);
        // scores[z] = Q[b,:,h,:] @ K[b,:,h,:]^T   (M=N=512, K=64), batched over z=b*H+h
        gemm(Qb, Kb, nullptr, SC, SEQL, SEQL, EHEAD, DMODEL, DMODEL, SEQL, 1, 0,
             BATCH * NHEADS, NHEADS, LD, EHEAD, LD, EHEAD,
             (long long)NHEADS * LSs, LSs);
        k_softmax<<<dim3(SEQL, BATCH * NHEADS), 256, 0, stream>>>(SC, tauv, dl, 0.125f, causal);
        // out[b,:,h,:] = A[z] @ V[b,:,h,:]   (M=512, N=64, K=512)
        gemm(SC, Vb, nullptr, T1, SEQL, EHEAD, SEQL, SEQL, DMODEL, DMODEL, 0, 0,
             BATCH * NHEADS, NHEADS, (long long)NHEADS * LSs, LSs, LD, EHEAD, LD, EHEAD);
        dense(T1, ow, ob, outY, NROWS, DMODEL, DMODEL, 0);
    };

    // ---- encoder ----
    k_embed<<<NROWS, 128, 0, stream>>>(xn, x_mark_enc, IN(40), IN(41), X);
    for (int l = 0; l < 2; ++l) {
        int p = 42 + 16 * l;
        attention(X, X, IN(p + 5), IN(p + 4), IN(p + 1), IN(p + 0),
                  IN(p + 7), IN(p + 6), IN(p + 3), IN(p + 2), 0, deltav, Y);
        k_add_ln<<<NROWS, 256, 0, stream>>>(X, Y, IN(p + 13), IN(p + 12), X);
        dense(X, IN(p + 9), IN(p + 8), FFH, NROWS, DFF, DMODEL, 1);     // fused GELU
        dense(FFH, IN(p + 11), IN(p + 10), Y, NROWS, DMODEL, DFF, 0);
        k_add_ln<<<NROWS, 256, 0, stream>>>(X, Y, IN(p + 15), IN(p + 14), X);
    }
    k_add_ln<<<NROWS, 256, 0, stream>>>(X, nullptr, IN(75), IN(74), ENC);

    // ---- decoder ----
    k_embed<<<NROWS, 128, 0, stream>>>(xdec, x_mark_dec, IN(4), IN(5), XD);
    // self-attention (causal, no delta)
    attention(XD, XD, IN(29), IN(28), IN(25), IN(24), IN(31), IN(30), IN(27), IN(26),
              1, nullptr, Y);
    k_add_ln<<<NROWS, 256, 0, stream>>>(XD, Y, IN(19), IN(18), XD);
    // cross-attention (tau + delta)
    attention(XD, ENC, IN(11), IN(10), IN(7), IN(6), IN(13), IN(12), IN(9), IN(8),
              0, deltav, Y);
    k_add_ln<<<NROWS, 256, 0, stream>>>(XD, Y, IN(21), IN(20), XD);
    dense(XD, IN(15), IN(14), FFH, NROWS, DFF, DMODEL, 1);
    dense(FFH, IN(17), IN(16), Y, NROWS, DMODEL, DFF, 0);
    k_add_ln<<<NROWS, 256, 0, stream>>>(XD, Y, IN(23), IN(22), XD);
    k_add_ln<<<NROWS, 256, 0, stream>>>(XD, nullptr, IN(33), IN(32), XD);

    // ---- projection + de-normalization, last PRED_LEN steps ----
    k_final<<<BATCH * PREDL, 32, 0, stream>>>(XD, IN(77), IN(76), meanv, stdv, (float*)d_out);
}
